// SkipTransformer_19284403159587
// MI455X (gfx1250) — matrix operand order repaired and checked by reading, not compiled, not run on hardware
//
#include <hip/hip_runtime.h>

// ---------------------------------------------------------------------------
// SkipTransformer forward for MI455X (gfx1250, wave32, WMMA).
// - All dense GEMMs: v_wmma_f32_16x16x32_bf16 (bf16 in, f32 accumulate).
// - LDS tiles stored in fragment-native chunk layout (16 K-values contiguous,
//   48B stride) -> B frags load as 2x ds_load_b128, bank-conflict-free.
// - 32x32 macro-tile per wave: 4 WMMA per K-step, A/B fragments reused.
// - Two-layer attention MLPs fused through LDS (hidden never hits HBM):
//   needs up to 240KB LDS/workgroup -> only possible on CDNA5 (320KB/WGP).
// ---------------------------------------------------------------------------

typedef __bf16 bf16;
typedef __attribute__((ext_vector_type(16))) __bf16 v16bf;
typedef __attribute__((ext_vector_type(8)))  __bf16 v8bf;
typedef __attribute__((ext_vector_type(8)))  float  v8f;

#define EPSBN 1e-5f
#define BATCH 8
#define NPTS  512
#define DIM   256
#define KNN_K 16

// bf16 elements reserved per 16-value chunk: 16 data + 8 pad = 48 bytes.
// 12-dword stride => chunk starts hit banks {0,4,...,60}: conflict-free b128s.
#define CHUNK_E 24

__device__ __forceinline__ v8f wmma_bf16(v16bf a, v16bf b, v8f c) {
  return __builtin_amdgcn_wmma_f32_16x16x32_bf16(
      /*neg_a=*/false, a, /*neg_b=*/false, b,
      /*c_mod=*/(short)0, c, /*reuse_a=*/false, /*reuse_b=*/false);
}

// A fragment (16x32 bf16) straight from row-major f32 W.
// lanes 0-15: M=lane, elems 0..15 = K+0..15
// lanes 16-31: M=lane-16, elems 0..15 = K+16..31
__device__ __forceinline__ v16bf load_a_frag_global(const float* __restrict__ W,
                                                    int ld, int m0, int k0, int lane) {
  const float* p = W + (long)(m0 + (lane & 15)) * ld + k0 + ((lane >> 4) << 4);
  v16bf a;
#pragma unroll
  for (int e = 0; e < 8; ++e) a[e] = (bf16)p[e];
#pragma unroll
  for (int e = 0; e < 8; ++e) a[8 + e] = (bf16)p[8 + e];
  return a;
}

// Stage a (Kdim x 32) column tile of X into LDS chunk layout.
// Chunk ch (= k16*32 + c) holds K values k16*16..+15 of column c, contiguous.
__device__ __forceinline__ void stage_tile(bf16* Bs, const float* __restrict__ Xb,
                                           int Kdim, int Ncols, int n0) {
  const int nchunk = (Kdim >> 4) * 32;
  for (int ch = threadIdx.x; ch < nchunk; ch += 256) {
    const int c = ch & 31;
    const int k0 = (ch >> 5) << 4;
    const float* src = Xb + (long)k0 * Ncols + n0 + c;
    v8bf lo, hi;
#pragma unroll
    for (int j = 0; j < 8; ++j) lo[j] = (bf16)src[(long)j * Ncols];
#pragma unroll
    for (int j = 0; j < 8; ++j) hi[j] = (bf16)src[(long)(8 + j) * Ncols];
    v8bf* dst = (v8bf*)(Bs + ch * CHUNK_E);
    dst[0] = lo;
    dst[1] = hi;
  }
}

// B fragment (32x16 bf16) from chunked LDS: two contiguous 16B loads.
// lane: col = n0 + (lane&15); elems j = K kb*32 + (lane>>4)*16 + j.
__device__ __forceinline__ v16bf load_b_frag(const bf16* Bs, int kb, int n0, int lane) {
  const bf16* p = Bs + ((kb << 6) + ((lane >> 4) << 5) + n0 + (lane & 15)) * CHUNK_E;
  v8bf lo = *(const v8bf*)p;
  v8bf hi = *(const v8bf*)(p + 8);
  v16bf b;
#pragma unroll
  for (int j = 0; j < 8; ++j) { b[j] = lo[j]; b[8 + j] = hi[j]; }
  return b;
}

// ---------------------------------------------------------------------------
// Generic GEMM: Y[b] (MxN) = W (MxK) @ X[b] (KxN) + bias [, ReLU] [, += Yold]
// 256 threads = 8 waves; block tile 256 rows x 32 cols; wave tile 32x32.
// ---------------------------------------------------------------------------
template <bool RELU, bool ADD>
__global__ __launch_bounds__(256) void gemm_kernel(
    const float* __restrict__ W, const float* __restrict__ bias,
    const float* __restrict__ X, float* __restrict__ Y,
    int M, int Kdim, int Ncols, long sXb, long sYb) {
  extern __shared__ __align__(16) char smem_raw[];
  bf16* Xs = (bf16*)smem_raw;

  const int lane = threadIdx.x & 31;
  const int wave = threadIdx.x >> 5;
  const int n0 = blockIdx.x * 32;
  const int b = blockIdx.z;
  const float* Xb = X + (long)b * sXb;
  float* Yb = Y + (long)b * sYb;

  stage_tile(Xs, Xb, Kdim, Ncols, n0);
  __syncthreads();

  const int m0 = blockIdx.y * 256 + wave * 32;
  __builtin_prefetch(W + (long)m0 * Kdim, 0, 1);
  v8f c00 = {}, c01 = {}, c10 = {}, c11 = {};
  for (int k = 0; k < Kdim; k += 32) {
    const int kb = k >> 5;
    v16bf a0 = load_a_frag_global(W, Kdim, m0, k, lane);
    v16bf a1 = load_a_frag_global(W, Kdim, m0 + 16, k, lane);
    v16bf b0 = load_b_frag(Xs, kb, 0, lane);
    v16bf b1 = load_b_frag(Xs, kb, 16, lane);
    c00 = wmma_bf16(a0, b0, c00);
    c01 = wmma_bf16(a0, b1, c01);
    c10 = wmma_bf16(a1, b0, c10);
    c11 = wmma_bf16(a1, b1, c11);
  }

  const int col = n0 + (lane & 15);
#pragma unroll
  for (int half = 0; half < 2; ++half) {
    const v8f& d0 = half ? c10 : c00;
    const v8f& d1 = half ? c11 : c01;
    const int rb = m0 + half * 16 + ((lane >> 4) << 3);
#pragma unroll
    for (int j = 0; j < 8; ++j) {
      int r = rb + j;
      float a0 = d0[j] + bias[r];
      float a1 = d1[j] + bias[r];
      if (RELU) { a0 = fmaxf(a0, 0.f); a1 = fmaxf(a1, 0.f); }
      float* p0 = Yb + (long)r * Ncols + col;
      float* p1 = p0 + 16;
      if (ADD) { a0 += *p0; a1 += *p1; }
      *p0 = a0;
      *p1 = a1;
    }
  }
}

// ---------------------------------------------------------------------------
// Fused 2-layer MLP: Y[b] = W2 @ relu(bn(W1 @ X[b] + b1)) + b2, 32-col tile.
// Hidden (H x 32, bf16, chunked) lives entirely in LDS.
// ---------------------------------------------------------------------------
__global__ __launch_bounds__(256) void fused_mlp_kernel(
    const float* __restrict__ W1, const float* __restrict__ b1,
    const float* __restrict__ g, const float* __restrict__ bet,
    const float* __restrict__ mu, const float* __restrict__ var,
    const float* __restrict__ W2, const float* __restrict__ b2,
    const float* __restrict__ X, float* __restrict__ Y,
    int H, int Cin, int Dout, int Ncols, long sXb, long sYb) {
  extern __shared__ __align__(16) char smem_raw[];
  bf16* Xs = (bf16*)smem_raw;                       // (Cin/16)*32 chunks
  bf16* Hs = Xs + (long)(Cin >> 4) * 32 * CHUNK_E;  // (H/16)*32 chunks

  const int lane = threadIdx.x & 31;
  const int wave = threadIdx.x >> 5;
  const int n0 = blockIdx.x * 32;
  const int b = blockIdx.y;
  const float* Xb = X + (long)b * sXb;
  float* Yb = Y + (long)b * sYb;

  stage_tile(Xs, Xb, Cin, Ncols, n0);
  __syncthreads();

  const int col0 = lane & 15;

  // Layer 1: hidden = relu(bn(W1 @ Xtile + b1)) -> LDS chunks (bf16)
  for (int rt = wave; rt < (H >> 5); rt += 8) {
    const int h0 = rt << 5;
    __builtin_prefetch(W1 + (long)(h0 + 32) * Cin, 0, 1);
    v8f c00 = {}, c01 = {}, c10 = {}, c11 = {};
    for (int k = 0; k < Cin; k += 32) {
      const int kb = k >> 5;
      v16bf a0 = load_a_frag_global(W1, Cin, h0, k, lane);
      v16bf a1 = load_a_frag_global(W1, Cin, h0 + 16, k, lane);
      v16bf b0 = load_b_frag(Xs, kb, 0, lane);
      v16bf b1 = load_b_frag(Xs, kb, 16, lane);
      c00 = wmma_bf16(a0, b0, c00);
      c01 = wmma_bf16(a0, b1, c01);
      c10 = wmma_bf16(a1, b0, c10);
      c11 = wmma_bf16(a1, b1, c11);
    }
#pragma unroll
    for (int half = 0; half < 2; ++half) {
      const v8f& d0 = half ? c10 : c00;
      const v8f& d1 = half ? c11 : c01;
      const int rb = h0 + half * 16 + ((lane >> 4) << 3);
#pragma unroll
      for (int j = 0; j < 8; ++j) {
        int r = rb + j;
        float s = g[r] * rsqrtf(var[r] + EPSBN);
        float t = bet[r] - mu[r] * s;
        float a0 = fmaxf((d0[j] + b1[r]) * s + t, 0.f);
        float a1 = fmaxf((d1[j] + b1[r]) * s + t, 0.f);
        const int chunk0 = ((r >> 4) << 5) + col0;
        Hs[chunk0 * CHUNK_E + (r & 15)] = (bf16)a0;
        Hs[(chunk0 + 16) * CHUNK_E + (r & 15)] = (bf16)a1;
      }
    }
  }
  __syncthreads();

  // Layer 2: Y = W2 @ hidden + b2
  for (int rt = wave; rt < (Dout >> 5); rt += 8) {
    const int m0 = rt << 5;
    __builtin_prefetch(W2 + (long)(m0 + 32) * H, 0, 1);
    v8f c00 = {}, c01 = {}, c10 = {}, c11 = {};
    for (int k = 0; k < H; k += 32) {
      const int kb = k >> 5;
      v16bf a0 = load_a_frag_global(W2, H, m0, k, lane);
      v16bf a1 = load_a_frag_global(W2, H, m0 + 16, k, lane);
      v16bf b0 = load_b_frag(Hs, kb, 0, lane);
      v16bf b1 = load_b_frag(Hs, kb, 16, lane);
      c00 = wmma_bf16(a0, b0, c00);
      c01 = wmma_bf16(a0, b1, c01);
      c10 = wmma_bf16(a1, b0, c10);
      c11 = wmma_bf16(a1, b1, c11);
    }
    const int col = n0 + col0;
#pragma unroll
    for (int half = 0; half < 2; ++half) {
      const v8f& d0 = half ? c10 : c00;
      const v8f& d1 = half ? c11 : c01;
      const int rb = m0 + half * 16 + ((lane >> 4) << 3);
#pragma unroll
      for (int j = 0; j < 8; ++j) {
        int r = rb + j;
        float* p = Yb + (long)r * Ncols + col;
        p[0] = d0[j] + b2[r];
        p[16] = d1[j] + b2[r];
      }
    }
  }
}

// ---------------------------------------------------------------------------
// Elementwise / gather / reduction kernels
// ---------------------------------------------------------------------------

__global__ __launch_bounds__(256) void concat_kernel(const float* __restrict__ key,
                                                     const float* __restrict__ query,
                                                     float* __restrict__ xcat) {
  long e = (long)blockIdx.x * 256 + threadIdx.x;  // 8*1024*512
  int n = e & 511;
  int c = (int)(e >> 9) & 1023;
  int b = (int)(e >> 19);
  float v = (c < 512) ? key[((long)b * 512 + c) * 512 + n]
                      : query[((long)b * 512 + (c - 512)) * 512 + n];
  xcat[e] = v;
}

// Brute-force KNN, K=16, include_self, tie-break by lower index (matches top_k).
__global__ __launch_bounds__(128) void knn_kernel(const float* __restrict__ pos,
                                                  int* __restrict__ idx) {
  __shared__ float xs[NPTS], ys[NPTS], zs[NPTS];
  const int b = blockIdx.y;
  const int n = blockIdx.x * 128 + threadIdx.x;
  const float* pb = pos + (long)b * 3 * NPTS;
  for (int i = threadIdx.x; i < NPTS; i += 128) {
    xs[i] = pb[i];
    ys[i] = pb[NPTS + i];
    zs[i] = pb[2 * NPTS + i];
  }
  __syncthreads();
  float qx = xs[n], qy = ys[n], qz = zs[n];
  float bd[KNN_K];
  int bi[KNN_K];
#pragma unroll
  for (int t = 0; t < KNN_K; ++t) { bd[t] = 3.4e38f; bi[t] = 0; }
  for (int m = 0; m < NPTS; ++m) {
    float dx = qx - xs[m], dy = qy - ys[m], dz = qz - zs[m];
    float d = dx * dx + dy * dy + dz * dz;
    if (d < bd[KNN_K - 1]) {
      int t = KNN_K - 1;
      while (t > 0 && bd[t - 1] > d) { bd[t] = bd[t - 1]; bi[t] = bi[t - 1]; --t; }
      bd[t] = d;
      bi[t] = m;
    }
  }
  int* op = idx + ((long)b * NPTS + n) * KNN_K;
#pragma unroll
  for (int t = 0; t < KNN_K; ++t) op[t] = bi[t];
}

// peh[b][h][n*K+k] = relu(bn(pm_w1 @ (pos[n]-pos[idx]) + pm_b1)), h in [0,64)
__global__ __launch_bounds__(256) void pe_hidden_kernel(
    const float* __restrict__ pos, const int* __restrict__ idx,
    const float* __restrict__ w1, const float* __restrict__ b1,
    const float* __restrict__ g, const float* __restrict__ bet,
    const float* __restrict__ mu, const float* __restrict__ var,
    float* __restrict__ peh) {
  long t = (long)blockIdx.x * 256 + threadIdx.x;  // B*N*K = 65536
  int n = (int)(t >> 4) & 511;
  int b = (int)(t >> 13);
  const float* pb = pos + (long)b * 3 * NPTS;
  int j = idx[t];
  float dx = pb[n] - pb[j];
  float dy = pb[NPTS + n] - pb[NPTS + j];
  float dz = pb[2 * NPTS + n] - pb[2 * NPTS + j];
  float* out = peh + (long)b * 64 * (NPTS * KNN_K) + (t & 8191);
  for (int h = 0; h < 64; ++h) {
    float v = w1[h * 3] * dx + w1[h * 3 + 1] * dy + w1[h * 3 + 2] * dz + b1[h];
    float s = g[h] * rsqrtf(var[h] + EPSBN);
    v = (v - mu[h]) * s + bet[h];
    out[(long)h * (NPTS * KNN_K)] = fmaxf(v, 0.f);
  }
}

// qkpe[b][d][n][k] = q[b][d][n] - kbuf[b][d][idx[b][n][k]] + pe[b][d][n][k]
__global__ __launch_bounds__(256) void qkpe_kernel(const float* __restrict__ q,
                                                   const float* __restrict__ kbuf,
                                                   const int* __restrict__ idx,
                                                   const float* __restrict__ pe,
                                                   float* __restrict__ qkpe) {
  long t = (long)blockIdx.x * 256 + threadIdx.x;  // B*DIM*N*K
  int kk = t & 15;
  int n = (int)(t >> 4) & 511;
  int d = (int)(t >> 13) & 255;
  int b = (int)(t >> 21);
  int j = idx[((long)b * NPTS + n) * KNN_K + kk];
  long rowq = ((long)b * DIM + d) * NPTS;
  qkpe[t] = q[rowq + n] - kbuf[rowq + j] + pe[t];
}

// xN[b][n][d*K+k] = qkpe[b][d][n*K+k]
__global__ __launch_bounds__(256) void xn_transpose_kernel(const float* __restrict__ qkpe,
                                                           float* __restrict__ xN) {
  long t = (long)blockIdx.x * 256 + threadIdx.x;  // B*N*DIM*K
  int col = t & 4095;
  int nc = (int)(t >> 12) & 511;
  int b = (int)(t >> 21);
  int d = col >> 4;
  int kk = col & 15;
  xN[t] = qkpe[((long)b * DIM + d) * (NPTS * KNN_K) + nc * KNN_K + kk];
}

// In-place softmax over contiguous groups of 16.
__global__ __launch_bounds__(256) void softmax16_kernel(float* __restrict__ buf) {
  long gidx = (long)blockIdx.x * 256 + threadIdx.x;
  float* p = buf + gidx * 16;
  float mx = p[0];
#pragma unroll
  for (int i = 1; i < 16; ++i) mx = fmaxf(mx, p[i]);
  float e[16];
  float s = 0.f;
#pragma unroll
  for (int i = 0; i < 16; ++i) { e[i] = __expf(p[i] - mx); s += e[i]; }
  float inv = 1.f / s;
#pragma unroll
  for (int i = 0; i < 16; ++i) p[i] = e[i] * inv;
}

// aggN[b][d][n] = sum_k aN[b][n][d*16+k] * (v[b][d][n] + pe[b][d][n*16+k])
__global__ __launch_bounds__(256) void aggN_kernel(const float* __restrict__ aN,
                                                   const float* __restrict__ v,
                                                   const float* __restrict__ pe,
                                                   float* __restrict__ aggN) {
  long t = (long)blockIdx.x * 256 + threadIdx.x;  // B*DIM*N
  int n = t & 511;
  int d = (int)(t >> 9) & 255;
  int b = (int)(t >> 17);
  float vv = v[t];
  const float* ap = aN + ((long)b * NPTS + n) * (DIM * KNN_K) + d * KNN_K;
  const float* pp = pe + ((long)b * DIM + d) * (NPTS * KNN_K) + n * KNN_K;
  float s = 0.f;
#pragma unroll
  for (int k = 0; k < KNN_K; ++k) s += ap[k] * (vv + pp[k]);
  aggN[t] = s;
}

// aggC[b][d][n] = sum_k aCh[b][d][n*16+k] * (v[b][d][n] + pe[b][d][n*16+k])
__global__ __launch_bounds__(256) void aggC_kernel(const float* __restrict__ aCh,
                                                   const float* __restrict__ v,
                                                   const float* __restrict__ pe,
                                                   float* __restrict__ aggC) {
  long t = (long)blockIdx.x * 256 + threadIdx.x;  // B*DIM*N
  float vv = v[t];
  const float* ap = aCh + t * KNN_K;
  const float* pp = pe + t * KNN_K;
  float s = 0.f;
#pragma unroll
  for (int k = 0; k < KNN_K; ++k) s += ap[k] * (vv + pp[k]);
  aggC[t] = s;
}

// ---------------------------------------------------------------------------
// Host-side launch sequence
// ---------------------------------------------------------------------------
extern "C" void kernel_launch(void* const* d_in, const int* in_sizes, int n_in,
                              void* d_out, int out_size, void* d_ws, size_t ws_size,
                              hipStream_t stream) {
  (void)in_sizes; (void)n_in; (void)out_size; (void)ws_size;

  const float* pos     = (const float*)d_in[0];
  const float* key     = (const float*)d_in[1];
  const float* query   = (const float*)d_in[2];
  const float* mlpv_w1 = (const float*)d_in[3];
  const float* mlpv_b1 = (const float*)d_in[4];
  const float* mlpv_w2 = (const float*)d_in[5];
  const float* mlpv_b2 = (const float*)d_in[6];
  const float* mlpv_ws = (const float*)d_in[7];
  const float* mlpv_bs = (const float*)d_in[8];
  const float* wk = (const float*)d_in[9];
  const float* bk = (const float*)d_in[10];
  const float* wq = (const float*)d_in[11];
  const float* bq = (const float*)d_in[12];
  const float* wv = (const float*)d_in[13];
  const float* bv = (const float*)d_in[14];
  const float* pm_w1 = (const float*)d_in[15];
  const float* pm_b1 = (const float*)d_in[16];
  const float* pm_g  = (const float*)d_in[17];
  const float* pm_be = (const float*)d_in[18];
  const float* pm_m  = (const float*)d_in[19];
  const float* pm_v  = (const float*)d_in[20];
  const float* pm_w2 = (const float*)d_in[21];
  const float* pm_b2 = (const float*)d_in[22];
  const float* am_w1 = (const float*)d_in[23];
  const float* am_b1 = (const float*)d_in[24];
  const float* am_g  = (const float*)d_in[25];
  const float* am_be = (const float*)d_in[26];
  const float* am_m  = (const float*)d_in[27];
  const float* am_v  = (const float*)d_in[28];
  const float* am_w2 = (const float*)d_in[29];
  const float* am_b2 = (const float*)d_in[30];
  const float* an_w1 = (const float*)d_in[31];
  const float* an_b1 = (const float*)d_in[32];
  const float* an_g  = (const float*)d_in[33];
  const float* an_be = (const float*)d_in[34];
  const float* an_m  = (const float*)d_in[35];
  const float* an_v  = (const float*)d_in[36];
  const float* an_w2 = (const float*)d_in[37];
  const float* an_b2 = (const float*)d_in[38];
  const float* we = (const float*)d_in[39];
  const float* be = (const float*)d_in[40];
  const float* wl = (const float*)d_in[41];
  const float* bl = (const float*)d_in[42];

  float* ws = (float*)d_ws;
  float* xcat  = ws;                       // 8*1024*512 = 4,194,304
  float* h1    = xcat  + 4194304;          // 2,097,152
  float* value = h1    + 2097152;          // 2,097,152
  float* kb    = value + 2097152;          // 1,048,576
  float* qb    = kb    + 1048576;          // 1,048,576
  float* vb    = qb    + 1048576;          // 1,048,576
  float* aggN  = vb    + 1048576;          // 1,048,576
  float* aggC  = aggN  + 1048576;          // 1,048,576
  float* ycat  = aggC  + 1048576;          // 4,194,304
  float* peh   = ycat  + 4194304;          // 4,194,304
  float* pe    = peh   + 4194304;          // 16,777,216
  float* qkpe  = pe    + 16777216;         // 16,777,216
  float* xN    = qkpe  + 16777216;         // 16,777,216 (reused for aCh)
  float* aN    = xN    + 16777216;         // 16,777,216
  int*   idx   = (int*)(aN + 16777216);    // 65,536 ints
  float* aCh   = xN;

  float* out = (float*)d_out;              // (B,512,512)
  const dim3 blk(256);
  const long sN  = 512L * 512;
  const long sD  = 256L * 512;
  const long sX  = 1024L * 512;
  const long sPE = 256L * 8192;
  const long sXN = 512L * 4096;
  const long sPH = 64L * 8192;
  // LDS bytes per K (chunked layout): (K/16)*32 chunks * 48B = K*96
  const int lds96_64 = 64 * 96, lds96_256 = 256 * 96, lds96_512 = 512 * 96,
            lds96_1024 = 1024 * 96;

  // 1) x = concat(key, query)
  concat_kernel<<<16384, blk, 0, stream>>>(key, query, xcat);

  // 2) MLP_Res: value = W2 @ relu(W1 @ x + b1) + b2 + (Ws @ x + bs)
  gemm_kernel<true, false><<<dim3(16, 2, 8), blk, lds96_1024, stream>>>(
      mlpv_w1, mlpv_b1, xcat, h1, 512, 1024, 512, sX, sN);
  gemm_kernel<false, false><<<dim3(16, 2, 8), blk, lds96_1024, stream>>>(
      mlpv_ws, mlpv_bs, xcat, value, 512, 1024, 512, sX, sN);
  gemm_kernel<false, true><<<dim3(16, 2, 8), blk, lds96_512, stream>>>(
      mlpv_w2, mlpv_b2, h1, value, 512, 512, 512, sN, sN);

  // 3) k, q, v projections (DIM=256)
  gemm_kernel<false, false><<<dim3(16, 1, 8), blk, lds96_512, stream>>>(
      wk, bk, key, kb, 256, 512, 512, sN, sD);
  gemm_kernel<false, false><<<dim3(16, 1, 8), blk, lds96_512, stream>>>(
      wq, bq, query, qb, 256, 512, 512, sN, sD);
  gemm_kernel<false, false><<<dim3(16, 1, 8), blk, lds96_512, stream>>>(
      wv, bv, value, vb, 256, 512, 512, sN, sD);

  // 4) KNN
  knn_kernel<<<dim3(4, 8), dim3(128), 0, stream>>>(pos, idx);

  // 5) pos_mlp: peh = relu(bn(pm_w1 @ pos_rel + b1)); pe = pm_w2 @ peh + b2
  pe_hidden_kernel<<<256, blk, 0, stream>>>(pos, idx, pm_w1, pm_b1,
                                            pm_g, pm_be, pm_m, pm_v, peh);
  gemm_kernel<false, false><<<dim3(256, 1, 8), blk, lds96_64, stream>>>(
      pm_w2, pm_b2, peh, pe, 256, 64, 8192, sPH, sPE);

  // 6) qk_rel + pe
  qkpe_kernel<<<65536, blk, 0, stream>>>(qb, kb, idx, pe, qkpe);

  // 7) N-branch: fused 512->2048->512 MLP over (B, N, DIM*K), then softmax
  xn_transpose_kernel<<<65536, blk, 0, stream>>>(qkpe, xN);
  fused_mlp_kernel<<<dim3(128, 8), blk, (512 + 2048) * 96, stream>>>(
      an_w1, an_b1, an_g, an_be, an_m, an_v, an_w2, an_b2,
      xN, aN, 2048, 512, 512, 4096, sXN, sXN);
  softmax16_kernel<<<4096, blk, 0, stream>>>(aN);

  // 8) channel-branch: fused 256->1024->256 MLP on qkpe, softmax (aCh reuses xN)
  fused_mlp_kernel<<<dim3(256, 8), blk, (256 + 1024) * 96, stream>>>(
      am_w1, am_b1, am_g, am_be, am_m, am_v, am_w2, am_b2,
      qkpe, aCh, 1024, 256, 256, 8192, sPE, sPE);
  softmax16_kernel<<<4096, blk, 0, stream>>>(aCh);

  // 9) attention aggregation
  aggN_kernel<<<4096, blk, 0, stream>>>(aN, vb, pe, aggN);
  aggC_kernel<<<4096, blk, 0, stream>>>(aCh, vb, pe, aggC);

  // 10) yN -> ycat[0:512]; y -> ycat[512:1024]
  gemm_kernel<false, false><<<dim3(16, 2, 8), blk, lds96_256, stream>>>(
      we, be, aggN, ycat, 512, 256, 512, sD, sX);
  gemm_kernel<false, false><<<dim3(16, 2, 8), blk, lds96_256, stream>>>(
      we, be, aggC, ycat + 512L * 512, 512, 256, 512, sD, sX);

  // 11) out = wl @ ycat + bl + identity(value)
  hipMemcpyAsync(out, value, (size_t)BATCH * 512 * 512 * sizeof(float),
                 hipMemcpyDeviceToDevice, stream);
  gemm_kernel<false, true><<<dim3(16, 2, 8), blk, lds96_1024, stream>>>(
      wl, bl, ycat, out, 512, 1024, 512, sX, sN);
}
